// GroupedQueryAttention_23407571763649
// MI455X (gfx1250) — matrix operand order repaired and checked by reading, not compiled, not run on hardware
//
#include <hip/hip_runtime.h>
#include <hip/hip_bf16.h>
#include <math.h>

// ---- problem constants (match reference) ----
#define Q_     1024
#define DIN_   4096
#define DOUT_  4096
#define H_     32
#define G_     8
#define HD_    128
#define PREV_  3072
#define KTOT_  4096   // PREV + Q
#define START_ 3072

typedef __attribute__((ext_vector_type(16))) __bf16    v16bf;
typedef __attribute__((ext_vector_type(8)))  __bf16    v8bf;
typedef __attribute__((ext_vector_type(8)))  float     v8f;
typedef __attribute__((ext_vector_type(8)))  unsigned  v8u;

__device__ __forceinline__ unsigned bfbits(float f) {
  unsigned u = __builtin_bit_cast(unsigned, f);
  return u + 0x7FFFu + ((u >> 16) & 1u);      // rne, result in high 16
}
__device__ __forceinline__ __bf16 f2bf(float f) {
  unsigned short h = (unsigned short)(bfbits(f) >> 16);
  return __builtin_bit_cast(__bf16, h);
}
__device__ __forceinline__ unsigned pack_bf16(float lo, float hi) {
  return (bfbits(lo) >> 16) | (bfbits(hi) & 0xFFFF0000u);
}

// A fragment: 16x32 bf16 tile of row-major A (ld in elements).
// Per lane: two contiguous 16B chunks (K kh..kh+7 and kh+16..kh+23).
__device__ __forceinline__ v16bf load_a_bf16(const __bf16* __restrict__ A, int ld, int lane) {
  const int row = lane & 15;
  const int kh  = (lane < 16) ? 0 : 8;
  const __bf16* p = A + (size_t)row * ld + kh;
  v8bf lo = *(const v8bf*)p;
  v8bf hi = *(const v8bf*)(p + 16);
  return __builtin_shufflevector(lo, hi, 0,1,2,3,4,5,6,7,8,9,10,11,12,13,14,15);
}

// B fragment: 32x16 bf16 tile, element (k,n) = Bt[n*strideN + k] (K contiguous).
// Per lane: two contiguous 16B chunks (K kh..kh+7 and kh+16..kh+23, kh = 0 or 8).
__device__ __forceinline__ v16bf load_b_bf16(const __bf16* __restrict__ Bt, int strideN, int lane) {
  const __bf16* p = Bt + (size_t)(lane & 15) * strideN + ((lane < 16) ? 0 : 8);
  v8bf lo = *(const v8bf*)p;
  v8bf hi = *(const v8bf*)(p + 16);
  return __builtin_shufflevector(lo, hi, 0,1,2,3,4,5,6,7,8,9,10,11,12,13,14,15);
}

// ---------------- fp32 -> bf16 bulk convert (4 elems/thread) ----------------
__global__ void cvt_bf16_kernel(const float* __restrict__ s, __bf16* __restrict__ d, int n4) {
  int i = blockIdx.x * blockDim.x + threadIdx.x;
  if (i >= n4) return;
  const float4 v = ((const float4*)s)[i];
  uint2 o;
  o.x = pack_bf16(v.x, v.y);
  o.y = pack_bf16(v.z, v.w);
  ((uint2*)d)[i] = o;
}

// ---------------- tiled transpose + convert: src fp32 [K,N] -> dst bf16 [N,K] ----------------
// block (32,8), grid (N/32, K/32, batch)
__global__ void transpose_cvt_kernel(const float* __restrict__ src, __bf16* __restrict__ dst,
                                     int K, int N, long sbatch, long dbatch) {
  __shared__ float t[32][33];
  const float* s = src + (size_t)blockIdx.z * sbatch;
  __bf16*      d = dst + (size_t)blockIdx.z * dbatch;
  const int n0 = blockIdx.x * 32, k0 = blockIdx.y * 32;
  const int tx = threadIdx.x, ty = threadIdx.y;
#pragma unroll
  for (int i = 0; i < 4; ++i)
    t[ty * 4 + i][tx] = s[(size_t)(k0 + ty * 4 + i) * N + n0 + tx];
  __syncthreads();
#pragma unroll
  for (int i = 0; i < 4; ++i)
    d[(size_t)(n0 + ty * 4 + i) * K + k0 + tx] = f2bf(t[tx][ty * 4 + i]);
}

// ---------------- GEMM: C[M,N] fp32 = A[M,K]bf16 x Bt[N,K]bf16 ----------------
// grid = (N/64, M/32), block = 32 (one wave -> 32x64 C tile, 8 WMMA / K-step)
__global__ void __launch_bounds__(32)
gemm_bf16_tn(const __bf16* __restrict__ A, const __bf16* __restrict__ Bt,
             float* __restrict__ C, int M, int N, int K) {
  const int lane = threadIdx.x;
  const int n0 = blockIdx.x * 64;
  const int m0 = blockIdx.y * 32;
  v8f acc[8] = {};
  const __bf16* A0 = A + (size_t)m0 * K;
  const __bf16* A1 = A + (size_t)(m0 + 16) * K;
  for (int k0 = 0; k0 < K; k0 += 32) {
    __builtin_prefetch(A0 + k0 + 512, 0, 1);   // -> global_prefetch_b8
    v16bf a0 = load_a_bf16(A0 + k0, K, lane);
    v16bf a1 = load_a_bf16(A1 + k0, K, lane);
#pragma unroll
    for (int j = 0; j < 4; ++j) {
      v16bf b = load_b_bf16(Bt + (size_t)(n0 + 16 * j) * K + k0, K, lane);
      acc[j]     = __builtin_amdgcn_wmma_f32_16x16x32_bf16(false, a0, false, b, (short)0, acc[j],     false, false);
      acc[4 + j] = __builtin_amdgcn_wmma_f32_16x16x32_bf16(false, a1, false, b, (short)0, acc[4 + j], false, false);
    }
  }
  // C layout: lane holds col = lane&15, rows = vgpr + (lane<16 ? 0 : 8)
  const int col = lane & 15;
  const int rb  = (lane < 16) ? 0 : 8;
#pragma unroll
  for (int half = 0; half < 2; ++half) {
    float* Cb = C + (size_t)(m0 + 16 * half) * N + n0;
#pragma unroll
    for (int i = 0; i < 8; ++i) {
      float* crow = Cb + (size_t)(rb + i) * N + col;
      crow[ 0] = acc[4 * half + 0][i];
      crow[16] = acc[4 * half + 1][i];
      crow[32] = acc[4 * half + 2][i];
      crow[48] = acc[4 * half + 3][i];
    }
  }
}

// ---------------- RoPE on q: fp32 in -> bf16 out (pairs d, d+64) ----------------
__global__ void rope_q_kernel(const float* __restrict__ q, __bf16* __restrict__ qb,
                              const float* __restrict__ cosb, const float* __restrict__ sinb) {
  int idx = blockIdx.x * blockDim.x + threadIdx.x; // s*2048 + h*64 + d
  if (idx >= Q_ * H_ * (HD_ / 2)) return;
  const int d = idx & 63;
  const int h = (idx >> 6) & 31;
  const int s = idx >> 11;
  const int pos = START_ + s;
  const size_t base = (size_t)s * DOUT_ + h * HD_;
  const float c0 = cosb[pos * HD_ + d], c1 = cosb[pos * HD_ + d + 64];
  const float s0 = sinb[pos * HD_ + d], s1 = sinb[pos * HD_ + d + 64];
  const float x1 = q[base + d], x2 = q[base + d + 64];
  qb[base + d]      = f2bf(x1 * c0 - x2 * s0);
  qb[base + d + 64] = f2bf(x2 * c1 + x1 * s1);
}

// ---------------- build keys/values fp32 outputs + bf16 keys copy ----------------
__global__ void build_kv_kernel(const float* __restrict__ kpro, const float* __restrict__ vpro,
                                const float* __restrict__ prev_k, const float* __restrict__ prev_v,
                                const float* __restrict__ cosb, const float* __restrict__ sinb,
                                float* __restrict__ keys, float* __restrict__ vals,
                                __bf16* __restrict__ keysb) {
  int idx = blockIdx.x * blockDim.x + threadIdx.x; // g*(4096*64) + pos*64 + d
  if (idx >= G_ * KTOT_ * (HD_ / 2)) return;
  const int d   = idx & 63;
  const int pos = (idx >> 6) & (KTOT_ - 1);
  const int g   = idx >> 18;
  const size_t ob = (size_t)g * KTOT_ * HD_ + (size_t)pos * HD_;
  float k1, k2, v1, v2;
  if (pos < PREV_) {
    const size_t pb = (size_t)g * PREV_ * HD_ + (size_t)pos * HD_;
    k1 = prev_k[pb + d]; k2 = prev_k[pb + d + 64];
    v1 = prev_v[pb + d]; v2 = prev_v[pb + d + 64];
  } else {
    const int s = pos - PREV_;
    const float* kp = kpro + (size_t)s * (G_ * HD_) + g * HD_;
    const float* vp = vpro + (size_t)s * (G_ * HD_) + g * HD_;
    const float c0 = cosb[pos * HD_ + d], c1 = cosb[pos * HD_ + d + 64];
    const float s0 = sinb[pos * HD_ + d], s1 = sinb[pos * HD_ + d + 64];
    const float x1 = kp[d], x2 = kp[d + 64];
    k1 = x1 * c0 - x2 * s0;
    k2 = x2 * c1 + x1 * s1;
    v1 = vp[d]; v2 = vp[d + 64];
  }
  keys[ob + d] = k1;  keys[ob + d + 64] = k2;
  vals[ob + d] = v1;  vals[ob + d + 64] = v2;
  keysb[ob + d] = f2bf(k1);  keysb[ob + d + 64] = f2bf(k2);
}

// ---------------- flash attention: one wave per (16-query tile, head) ----------------
// keysb: bf16 [g][key][d]; valsT: bf16 [g][d][key]; qb: bf16 [s][H*HD]; ctxb out bf16 [s][H*HD]
__global__ void __launch_bounds__(32)
attn_kernel(const __bf16* __restrict__ qb, const __bf16* __restrict__ keysb,
            const __bf16* __restrict__ valsT, __bf16* __restrict__ ctxb) {
  const int lane = threadIdx.x;
  const int qt = blockIdx.x;          // 0..63
  const int h  = blockIdx.y;          // 0..31
  const int g  = h >> 2;              // GQA group (gs = H/G = 4)
  const int q0 = qt * 16;

  __shared__ float Sm[16][33];                     // raw score tile (transposed via LDS)
  __shared__ __align__(32) unsigned Pp[32][8];     // P as packed bf16 pairs, per consumer lane
  __shared__ float mrow[16], lrow[16], crow[16];

  if (lane < 16) { mrow[lane] = -3.0e38f; lrow[lane] = 0.f; crow[lane] = 0.f; }
  __syncthreads();

  // Q tile 16x128 as four 16x32 A fragments (contiguous b128 loads)
  const __bf16* Qb = qb + (size_t)q0 * DOUT_ + h * HD_;
  v16bf aq0 = load_a_bf16(Qb +  0, DOUT_, lane);
  v16bf aq1 = load_a_bf16(Qb + 32, DOUT_, lane);
  v16bf aq2 = load_a_bf16(Qb + 64, DOUT_, lane);
  v16bf aq3 = load_a_bf16(Qb + 96, DOUT_, lane);

  v8f acc[8] = {};                                 // ctx 16x128 as eight 16x16 C tiles

  const __bf16* Kg = keysb + (size_t)g * KTOT_ * HD_;
  const __bf16* Vg = valsT + (size_t)g * HD_ * KTOT_;

  const int kend = ((START_ + q0 + 16) + 31) & ~31;   // causal key bound, 32-aligned
  const float scale = 0.08838834764831845f;           // 1/sqrt(128)

  for (int kb = 0; kb < kend; kb += 32) {
    // ---- scores: S(16x32) = Q(16x128) . K^T ----
    v8f sc0 = {}, sc1 = {};
    const __bf16* Kb0 = Kg + (size_t)kb * HD_;
#pragma unroll
    for (int kc = 0; kc < 4; ++kc) {
      // B element (K=d, N=key) = keysb[key][d] -> strideN = HD, K contiguous
      v16bf b0 = load_b_bf16(Kb0 + kc * 32,            HD_, lane);
      v16bf b1 = load_b_bf16(Kb0 + 16 * HD_ + kc * 32, HD_, lane);
      v16bf a  = (kc == 0) ? aq0 : (kc == 1) ? aq1 : (kc == 2) ? aq2 : aq3;
      sc0 = __builtin_amdgcn_wmma_f32_16x16x32_bf16(false, a, false, b0, (short)0, sc0, false, false);
      sc1 = __builtin_amdgcn_wmma_f32_16x16x32_bf16(false, a, false, b1, (short)0, sc1, false, false);
    }
    // spill score tile to LDS (C-layout -> row-major transpose)
    {
      const int col = lane & 15;
      const int rb  = (lane < 16) ? 0 : 8;
#pragma unroll
      for (int i = 0; i < 8; ++i) {
        Sm[rb + i][col]      = sc0[i];
        Sm[rb + i][col + 16] = sc1[i];
      }
    }
    __syncthreads();
    // ---- online softmax, one row per lane<16; pack P for consumer lanes ----
    if (lane < 16) {
      const int r = lane;
      const int qpos = START_ + q0 + r;
      float pv[32];
      const float mold = mrow[r];
      float mnew = mold;
#pragma unroll
      for (int c = 0; c < 32; ++c) {
        float sv = Sm[r][c] * scale;
        if (kb + c > qpos) sv = -3.0e38f;            // causal mask
        pv[c] = sv;
        mnew = fmaxf(mnew, sv);
      }
      const float cor = __expf(mold - mnew);
      float sum = 0.f;
#pragma unroll
      for (int c = 0; c < 32; ++c) {
        pv[c] = __expf(pv[c] - mnew);
        sum += pv[c];
      }
      mrow[r] = mnew;
      lrow[r] = lrow[r] * cor + sum;
      crow[r] = cor;
      // pack into A-fragment order for consumer lanes r (kh=0) and r+16 (kh=8)
      v8u lo8, hi8;
#pragma unroll
      for (int v = 0; v < 8; ++v) {
        const int k = (v < 4) ? (2 * v) : (16 + 2 * (v - 4));
        lo8[v] = pack_bf16(pv[k],     pv[k + 1]);
        hi8[v] = pack_bf16(pv[k + 8], pv[k + 9]);
      }
      *(v8u*)(&Pp[r][0])      = lo8;
      *(v8u*)(&Pp[r + 16][0]) = hi8;
    }
    __syncthreads();
    // rescale running context by per-row correction
    {
      const int rb = (lane < 16) ? 0 : 8;
#pragma unroll
      for (int i = 0; i < 8; ++i) {
        const float cf = crow[rb + i];
#pragma unroll
        for (int n = 0; n < 8; ++n) acc[n][i] *= cf;
      }
    }
    // ---- ctx += P(16x32) . V(32x128) ----
    v16bf pf = __builtin_bit_cast(v16bf, *(const v8u*)(&Pp[lane][0]));
    const __bf16* Vb = Vg + kb;       // element (K=key, N=d) = valsT[d][kb+key] -> strideN = KTOT
#pragma unroll
    for (int n = 0; n < 8; ++n) {
      v16bf bv = load_b_bf16(Vb + (size_t)(n * 16) * KTOT_, KTOT_, lane);
      acc[n] = __builtin_amdgcn_wmma_f32_16x16x32_bf16(false, pf, false, bv, (short)0, acc[n], false, false);
    }
    __syncthreads();
  }

  // epilogue: divide by softmax denominator, store bf16 ctx[s, h*128 + d]
  const int col = lane & 15;
  const int rb  = (lane < 16) ? 0 : 8;
  __bf16* Cb = ctxb + (size_t)q0 * DOUT_ + h * HD_;
#pragma unroll
  for (int i = 0; i < 8; ++i) {
    const float inv = 1.f / lrow[rb + i];
#pragma unroll
    for (int n = 0; n < 8; ++n)
      Cb[(size_t)(rb + i) * DOUT_ + n * 16 + col] = f2bf(acc[n][i] * inv);
  }
}

// ---------------- host-side launcher ----------------
extern "C" void kernel_launch(void* const* d_in, const int* in_sizes, int n_in,
                              void* d_out, int out_size, void* d_ws, size_t ws_size,
                              hipStream_t stream) {
  (void)in_sizes; (void)n_in; (void)out_size; (void)ws_size;
  const float* x      = (const float*)d_in[0];
  // d_in[1] = mask (unused; causality computed analytically)
  const float* cosb   = (const float*)d_in[2];
  const float* sinb   = (const float*)d_in[3];
  // d_in[4] = start_pos scalar (compile-time constant START_)
  const float* prev_k = (const float*)d_in[5];
  const float* prev_v = (const float*)d_in[6];
  const float* Wq     = (const float*)d_in[7];
  const float* Wk     = (const float*)d_in[8];
  const float* Wv     = (const float*)d_in[9];
  const float* Wo     = (const float*)d_in[10];

  float* out  = (float*)d_out;                          // (1024, 4096)
  float* keys = out  + (size_t)Q_ * DOUT_;              // (8, 4096, 128)
  float* vals = keys + (size_t)G_ * KTOT_ * HD_;        // (8, 4096, 128)

  // ---- workspace carve (bytes) ----
  char* w = (char*)d_ws;
  float*  qbuf  = (float*)w;  w += (size_t)Q_ * DOUT_      * 4;  // q proj fp32      16MB
  float*  kpro  = (float*)w;  w += (size_t)Q_ * G_ * HD_   * 4;  // k proj fp32       4MB
  float*  vpro  = (float*)w;  w += (size_t)Q_ * G_ * HD_   * 4;  // v proj fp32       4MB
  __bf16* xb    = (__bf16*)w; w += (size_t)Q_ * DIN_       * 2;  // x bf16            8MB
  __bf16* WqT   = (__bf16*)w; w += (size_t)DOUT_ * DIN_    * 2;  // Wq^T bf16        32MB
  __bf16* WkT   = (__bf16*)w; w += (size_t)(G_*HD_) * DIN_ * 2;  // Wk^T bf16         8MB
  __bf16* WvT   = (__bf16*)w; w += (size_t)(G_*HD_) * DIN_ * 2;  // Wv^T bf16         8MB
  __bf16* WoT   = (__bf16*)w; w += (size_t)DOUT_ * DOUT_   * 2;  // Wo^T bf16        32MB
  __bf16* qbb   = (__bf16*)w; w += (size_t)Q_ * DOUT_      * 2;  // roped q bf16      8MB
  __bf16* keysb = (__bf16*)w; w += (size_t)G_ * KTOT_ * HD_* 2;  // keys bf16         8MB
  __bf16* valsT = (__bf16*)w; w += (size_t)G_ * HD_ * KTOT_* 2;  // vals^T bf16       8MB
  __bf16* ctxb  = (__bf16*)w; w += (size_t)Q_ * DOUT_      * 2;  // ctx bf16          8MB

  const dim3 wave(32);
  const dim3 tblk(32, 8);

  // ---- pre-pass: bf16 conversions / weight transposes ----
  {
    const int n4 = (Q_ * DIN_) / 4;
    cvt_bf16_kernel<<<(n4 + 255) / 256, 256, 0, stream>>>(x, xb, n4);
  }
  transpose_cvt_kernel<<<dim3(DOUT_/32, DIN_/32, 1),      tblk, 0, stream>>>(Wq, WqT, DIN_, DOUT_,   0, 0);
  transpose_cvt_kernel<<<dim3((G_*HD_)/32, DIN_/32, 1),   tblk, 0, stream>>>(Wk, WkT, DIN_, G_*HD_,  0, 0);
  transpose_cvt_kernel<<<dim3((G_*HD_)/32, DIN_/32, 1),   tblk, 0, stream>>>(Wv, WvT, DIN_, G_*HD_,  0, 0);
  transpose_cvt_kernel<<<dim3(DOUT_/32, DOUT_/32, 1),     tblk, 0, stream>>>(Wo, WoT, DOUT_, DOUT_,  0, 0);

  // ---- projections: q = x@Wq, k = x@Wk, v = x@Wv ----
  gemm_bf16_tn<<<dim3(DOUT_/64,    Q_/32), wave, 0, stream>>>(xb, WqT, qbuf, Q_, DOUT_,   DIN_);
  gemm_bf16_tn<<<dim3((G_*HD_)/64, Q_/32), wave, 0, stream>>>(xb, WkT, kpro, Q_, G_*HD_,  DIN_);
  gemm_bf16_tn<<<dim3((G_*HD_)/64, Q_/32), wave, 0, stream>>>(xb, WvT, vpro, Q_, G_*HD_,  DIN_);

  // ---- RoPE q (fp32 -> bf16) ----
  {
    const int n = Q_ * H_ * (HD_ / 2);
    rope_q_kernel<<<(n + 255) / 256, 256, 0, stream>>>(qbuf, qbb, cosb, sinb);
  }
  // ---- keys/values outputs (fp32) + bf16 keys ----
  {
    const int n = G_ * KTOT_ * (HD_ / 2);
    build_kv_kernel<<<(n + 255) / 256, 256, 0, stream>>>(kpro, vpro, prev_k, prev_v,
                                                         cosb, sinb, keys, vals, keysb);
  }
  // ---- vals (fp32, [g][pos][d]) -> valsT (bf16, [g][d][pos]) ----
  transpose_cvt_kernel<<<dim3(HD_/32, KTOT_/32, G_), tblk, 0, stream>>>(
      vals, valsT, KTOT_, HD_, (long)KTOT_ * HD_, (long)HD_ * KTOT_);

  // ---- flash attention ----
  attn_kernel<<<dim3(Q_/16, H_), wave, 0, stream>>>(qbb, keysb, valsT, ctxb);

  // ---- output projection: out = ctx @ Wo ----
  gemm_bf16_tn<<<dim3(DOUT_/64, Q_/32), wave, 0, stream>>>(ctxb, WoT, out, Q_, DOUT_, DIN_);
}
